// MultiHeadAttention_53420803228274
// MI455X (gfx1250) — compile-verified, matrix operand order checked by reading, not run on hardware
//
#include <hip/hip_runtime.h>

// ---------------------------------------------------------------------------
// MI455X (gfx1250) multi-head attention forward.
// All matmuls via V_WMMA_F32_16X16X32_BF16 (bf16 in, f32 acc).
// GEMM B tiles staged with global_load_async_to_lds_b128 (ASYNCcnt);
// attention K/V tiles staged with the Tensor Data Mover (TENSORcnt).
// ---------------------------------------------------------------------------

typedef __attribute__((ext_vector_type(4)))  float  v4f;
typedef __attribute__((ext_vector_type(8)))  float  v8f;
typedef __attribute__((ext_vector_type(8)))  __bf16 v8bf;
typedef __attribute__((ext_vector_type(16))) __bf16 v16bf;
typedef __attribute__((ext_vector_type(4)))  unsigned int u32x4;
typedef __attribute__((ext_vector_type(4)))  int i32x4;
typedef __attribute__((ext_vector_type(8)))  int i32x8;

constexpr int Hc  = 2048;  // hidden
constexpr int NHc = 16;    // heads
constexpr int HDc = 128;   // head dim
constexpr int Bc  = 2;     // batch
constexpr int Sc  = 2048;  // seq len

#define DEVFN static __device__ __forceinline__

DEVFN v8f wmma_bf16(v16bf a, v16bf b, v8f c) {
  return __builtin_amdgcn_wmma_f32_16x16x32_bf16(
      false, a, false, b, (short)0, c, false, false);
}

DEVFN uint32_t lds_off(const void* p) {
  // Flat LDS addresses carry the LDS byte offset in the low 32 bits.
  return (uint32_t)(uintptr_t)p;
}

// One wave instruction: every lane copies 16 bytes global -> LDS (ASYNCcnt).
DEVFN void async_b128(uint32_t lds, uint64_t gaddr) {
  asm volatile("global_load_async_to_lds_b128 %0, %1, off"
               :: "v"(lds), "v"(gaddr) : "memory");
}
DEVFN void wait_async0() { asm volatile("s_wait_asynccnt 0x0" ::: "memory"); }

// Tensor Data Mover: 2D tile (tile_w elems/row x tile_h rows, bf16) from
// global (row stride = stride_elems) into LDS, rows packed contiguously.
// D# bitfields per CDNA5 ISA ch.8 (group0: count|lds|gaddr|type=2;
// group1: data_size=2B, tensor/tile dims, dim0 stride).
#if __has_include(<hip/amd_detail/amd_gfx1250_TDM.h>)
#define TDM_SIX_ARGS 1
#else
#define TDM_SIX_ARGS 0
#endif

DEVFN void tdm_load_2d(uint32_t lds_addr, uint64_t gaddr,
                       uint32_t tile_w, uint32_t tile_h,
                       uint32_t stride_elems) {
  u32x4 g0;
  g0[0] = 1u;                                             // count=1, user D#
  g0[1] = lds_addr;                                       // lds_addr[31:0]
  g0[2] = (uint32_t)(gaddr & 0xffffffffu);                // global_addr lo
  g0[3] = (uint32_t)((gaddr >> 32) & 0x1ffffffu)          // global_addr hi
        | (2u << 30);                                     // type = image
  i32x8 g1;
  g1[0] = 0x10000;                                        // data_size=1 (2B)
  g1[1] = (int)((tile_w & 0xffffu) << 16);                // tensor_dim0 lo16
  g1[2] = (int)(((tile_w >> 16) & 0xffffu)                // tensor_dim0 hi16
        | ((tile_h & 0xffffu) << 16));                    // tensor_dim1 lo16
  g1[3] = (int)(((tile_h >> 16) & 0xffffu)                // tensor_dim1 hi16
        | ((tile_w & 0xffffu) << 16));                    // tile_dim0
  g1[4] = (int)(tile_h & 0xffffu);                        // tile_dim1 (dim2=0)
  g1[5] = (int)stride_elems;                              // dim0_stride lo32
  g1[6] = 0;
  g1[7] = 0;
  const i32x4 gz = {0, 0, 0, 0};
#if TDM_SIX_ARGS
  const i32x8 gz8 = {0, 0, 0, 0, 0, 0, 0, 0};
  __builtin_amdgcn_tensor_load_to_lds(g0, g1, gz, gz, gz8, 0);
#else
  __builtin_amdgcn_tensor_load_to_lds(g0, g1, gz, gz, 0);
#endif
}

// 16x32 bf16 fragment from a row-major bf16 source (global or LDS), ld in
// elements. Serves A fragments (row = M) and B fragments (row = N when the
// source is K-contiguous per output column). ISA 16-bit 16x32 striping:
// lanes 0-15 -> K in [k0,k0+8) U [k0+16,k0+24); lanes 16-31 -> +8.
DEVFN v16bf load_frag(const __bf16* base, int ld, int row0, int k0) {
  const int lane = threadIdx.x & 31;
  const __bf16* p = base + (size_t)(row0 + (lane & 15)) * ld
                         + k0 + ((lane >> 4) << 3);
  v8bf lo = *reinterpret_cast<const v8bf*>(p);
  v8bf hi = *reinterpret_cast<const v8bf*>(p + 16);
  v16bf r;
#pragma unroll
  for (int i = 0; i < 8; ++i) { r[i] = lo[i]; r[8 + i] = hi[i]; }
  return r;
}

// ---------------------------------------------------------------------------
// Pack kernels (one-time f32 -> bf16 conversions).
// ---------------------------------------------------------------------------
__global__ void pack_bf16_kernel(const float* __restrict__ in,
                                 __bf16* __restrict__ out, size_t n) {
  size_t j = ((size_t)blockIdx.x * blockDim.x + threadIdx.x) * 4;
  if (j >= n) return;
  v4f x = *reinterpret_cast<const v4f*>(in + j);
#pragma unroll
  for (int k = 0; k < 4; ++k) out[j + k] = (__bf16)x[k];
}

// RoPE fused with bf16 pack: in f32 [B,NH,S,HD] -> out bf16 same layout.
__global__ void rope_pack_kernel(const float* __restrict__ in,
                                 __bf16* __restrict__ out,
                                 int total_pairs, float mul) {
  int idx = blockIdx.x * blockDim.x + threadIdx.x;
  if (idx >= total_pairs) return;
  const int r = idx >> 6;      // row in [B*NH*S)
  const int d = idx & 63;
  const int s = r & (Sc - 1);
  const float inv = __expf(-(float)d * (9.210340371976184f / 64.0f));
  const float ang = (float)s * inv;
  const float c = cosf(ang), si = sinf(ang);
  const float* p = in + (size_t)r * HDc;
  const float x0 = p[d], x1 = p[d + 64];
  out[(size_t)r * HDc + d]      = (__bf16)((x0 * c - x1 * si) * mul);
  out[(size_t)r * HDc + d + 64] = (__bf16)((x1 * c + x0 * si) * mul);
}

// V transpose pack: f32 [BH][S][HD] -> bf16 [BH][HD][S] (output-coalesced).
__global__ void packT_kernel(const float* __restrict__ V,
                             __bf16* __restrict__ VT, size_t n) {
  size_t i = (size_t)blockIdx.x * blockDim.x + threadIdx.x;
  if (i >= n) return;
  const int s  = (int)(i & (Sc - 1));
  const size_t rest = i >> 11;          // / Sc
  const int d  = (int)(rest & (HDc - 1));
  const int bh = (int)(rest >> 7);      // / HDc
  VT[i] = (__bf16)V[((size_t)bh * Sc + s) * HDc + d];
}

// ---------------------------------------------------------------------------
// GEMM: Y = X[M,K](bf16) * W[N,K](bf16)^T + bias. 128 thr (4 waves),
// 64x128 output tile, 8 WMMAs per wave per K-step. B tile (shared by all
// waves) double-buffered in LDS via async copy; peeled loop (no in-loop
// staging branch). qkv_layout==1 scatters f32 to [B,NH,S,HD].
// ---------------------------------------------------------------------------
__global__ __launch_bounds__(128) void gemm_bf16_kernel(
    const __bf16* __restrict__ X, const __bf16* __restrict__ W,
    const float* __restrict__ bias, float* __restrict__ Y,
    int M, int N, int K, int qkv_layout) {
  __shared__ __align__(16) __bf16 wtile[2][128 * 32];  // [n][k], 2x8KB

  const int w    = threadIdx.x >> 5;
  const int lane = threadIdx.x & 31;
  const int m0 = blockIdx.y * 64 + w * 16;
  const int n0 = blockIdx.x * 128;
  const int srow = threadIdx.x;  // staging: one 32-elem row (64B) per thread

  v8f acc[8];
  const v8f vz = {0.f, 0.f, 0.f, 0.f, 0.f, 0.f, 0.f, 0.f};
#pragma unroll
  for (int c = 0; c < 8; ++c) acc[c] = vz;

  auto stage = [&](int k0, int bufi) {
    uint64_t g = (uint64_t)(uintptr_t)(W + (size_t)(n0 + srow) * K + k0);
    uint32_t l = lds_off(&wtile[bufi][srow * 32]);
    async_b128(l,      g);
    async_b128(l + 16, g + 16);
    async_b128(l + 32, g + 32);
    async_b128(l + 48, g + 48);
  };
  auto compute = [&](int k0, int bufi) {
    v16bf a = load_frag(X, K, m0, k0);
    const __bf16* wt = &wtile[bufi][0];
#pragma unroll
    for (int c = 0; c < 8; ++c) {
      v16bf b = load_frag(wt, 32, c * 16, 0);
      acc[c] = wmma_bf16(a, b, acc[c]);
    }
  };

  stage(0, 0);
  wait_async0();
  __syncthreads();

  int buf = 0;
  const int nsteps = K / 32;  // >= 2
  for (int st = 0; st < nsteps - 1; ++st) {
    stage((st + 1) * 32, buf ^ 1);
    __builtin_prefetch(X + (size_t)(m0 + (lane & 15)) * K + (st + 1) * 32,
                       0, 3);
    compute(st * 32, buf);
    wait_async0();
    __syncthreads();
    buf ^= 1;
  }
  compute((nsteps - 1) * 32, buf);

  const int hi  = lane >> 4;
  const int col = lane & 15;
#pragma unroll
  for (int c = 0; c < 8; ++c) {
    const int n = n0 + c * 16 + col;
    const float bv = bias[n];
#pragma unroll
    for (int j = 0; j < 8; ++j) {
      const int m = m0 + j + hi * 8;
      const float val = acc[c][j] + bv;
      if (!qkv_layout) {
        Y[(size_t)m * N + n] = val;
      } else {
        const int b = m / Sc, s = m - b * Sc;
        const int h = n / HDc, d = n - h * HDc;
        Y[(((size_t)b * NHc + h) * Sc + s) * HDc + d] = val;
      }
    }
  }
}

// ---------------------------------------------------------------------------
// Flash attention. Grid (S/64, B*NH), 128 thr (4 waves). Wave w owns Q rows
// [q0+16w, +16). K and V^T tiles staged by wave 0 through the Tensor Data
// Mover (double-buffered, TENSORcnt). P transposed via per-wave LDS.
// Output written as bf16 [B,S,H] for the output-projection GEMM.
// ---------------------------------------------------------------------------
__global__ __launch_bounds__(128) void attn_kernel(
    const __bf16* __restrict__ Qg, const __bf16* __restrict__ Kg,
    const __bf16* __restrict__ VTg, __bf16* __restrict__ Ag) {
  __shared__ __align__(16) __bf16 ktile[2][32 * 128];  // [kk][d]   2x8KB
  __shared__ __align__(16) __bf16 vtile[2][128 * 32];  // [d][kk]   2x8KB
  __shared__ __align__(16) __bf16 pbuf[4][16 * 32];    // per-wave P  4KB

  const int w    = threadIdx.x >> 5;
  const int lane = threadIdx.x & 31;
  const int bh   = blockIdx.y;
  const int b    = bh >> 4;          // NH == 16
  const int head = bh & 15;
  const int q0   = blockIdx.x * 64;
  const int qr0  = q0 + w * 16;

  const __bf16* qp = Qg  + (size_t)bh * Sc * HDc;  // [s][d], scale folded
  const __bf16* kp = Kg  + (size_t)bh * Sc * HDc;  // [s][d]
  const __bf16* vp = VTg + (size_t)bh * HDc * Sc;  // [d][s]

  v16bf aq[4];
#pragma unroll
  for (int dc = 0; dc < 4; ++dc) aq[dc] = load_frag(qp, HDc, qr0, dc * 32);

  const v8f vz = {0.f, 0.f, 0.f, 0.f, 0.f, 0.f, 0.f, 0.f};
  v8f oacc[8];
#pragma unroll
  for (int c = 0; c < 8; ++c) oacc[c] = vz;
  float m_i[8], l_i[8];
#pragma unroll
  for (int j = 0; j < 8; ++j) { m_i[j] = -1e30f; l_i[j] = 0.0f; }

  const int hi  = lane >> 4;
  const int col = lane & 15;

  auto stage = [&](int kk0, int bufi) {
    if (threadIdx.x < 32) {  // wave 0 drives the TDM
      tdm_load_2d(lds_off(&ktile[bufi][0]),
                  (uint64_t)(uintptr_t)(kp + (size_t)kk0 * HDc),
                  /*tile_w=*/HDc, /*tile_h=*/32, /*stride=*/HDc);
      tdm_load_2d(lds_off(&vtile[bufi][0]),
                  (uint64_t)(uintptr_t)(vp + kk0),
                  /*tile_w=*/32, /*tile_h=*/HDc, /*stride=*/Sc);
    }
  };
  auto wait_stage = [&]() {
    if (threadIdx.x < 32) __builtin_amdgcn_s_wait_tensorcnt((short)0);
    __syncthreads();
  };

  auto process = [&](int kk0, int bufi) {
    // scores: two 16x16 blocks over 4 K=32 chunks, B fragments from LDS.
    v8f sc0 = vz, sc1 = vz;
    const __bf16* kl = &ktile[bufi][0];
#pragma unroll
    for (int dc = 0; dc < 4; ++dc) {
      v16bf b0 = load_frag(kl, HDc, 0,  dc * 32);
      v16bf b1 = load_frag(kl, HDc, 16, dc * 32);
      sc0 = wmma_bf16(aq[dc], b0, sc0);
      sc1 = wmma_bf16(aq[dc], b1, sc1);
    }

    const bool need_mask = (kk0 + 31 > qr0);  // wave-uniform
    float f[8];
#pragma unroll
    for (int j = 0; j < 8; ++j) {
      float s0 = sc0[j], s1 = sc1[j];
      if (need_mask) {
        const int qrow = qr0 + j + hi * 8;
        if (kk0 + col > qrow)      s0 = -1e30f;
        if (kk0 + 16 + col > qrow) s1 = -1e30f;
      }
      float t = fmaxf(s0, s1);
      t = fmaxf(t, __shfl_xor(t, 8, 16));
      t = fmaxf(t, __shfl_xor(t, 4, 16));
      t = fmaxf(t, __shfl_xor(t, 2, 16));
      t = fmaxf(t, __shfl_xor(t, 1, 16));
      const float mo = m_i[j];
      const float mn = fmaxf(mo, t);
      const float p0 = __expf(s0 - mn);
      const float p1 = __expf(s1 - mn);
      float rs = p0 + p1;
      rs += __shfl_xor(rs, 8, 16);
      rs += __shfl_xor(rs, 4, 16);
      rs += __shfl_xor(rs, 2, 16);
      rs += __shfl_xor(rs, 1, 16);
      const float fac = __expf(mo - mn);
      l_i[j] = l_i[j] * fac + rs;
      m_i[j] = mn;
      f[j] = fac;
      pbuf[w][(j + hi * 8) * 32 + col]      = (__bf16)p0;
      pbuf[w][(j + hi * 8) * 32 + 16 + col] = (__bf16)p1;
    }

#pragma unroll
    for (int c = 0; c < 8; ++c)
#pragma unroll
      for (int j = 0; j < 8; ++j) oacc[c][j] *= f[j];

    __builtin_amdgcn_sched_barrier(0);  // keep P stores before P reads

    v16bf ap = load_frag(&pbuf[w][0], 32, 0, 0);
    const __bf16* vl = &vtile[bufi][0];
#pragma unroll
    for (int c = 0; c < 8; ++c) {
      v16bf bv = load_frag(vl, 32, c * 16, 0);
      oacc[c] = wmma_bf16(ap, bv, oacc[c]);
    }
  };

  const int nkt = q0 / 32 + 2;  // causal bound (uniform per block, >= 2)

  stage(0, 0);
  wait_stage();

  int buf = 0;
  for (int kt = 0; kt < nkt - 1; ++kt) {
    stage((kt + 1) * 32, buf ^ 1);
    process(kt * 32, buf);
    wait_stage();
    buf ^= 1;
  }
  process((nkt - 1) * 32, buf);

  // Epilogue: O * (1/l) -> bf16 [B, S, H].
  float rl[8];
#pragma unroll
  for (int j = 0; j < 8; ++j) rl[j] = 1.0f / l_i[j];
#pragma unroll
  for (int c = 0; c < 8; ++c) {
#pragma unroll
    for (int j = 0; j < 8; ++j) {
      const int s = qr0 + j + hi * 8;
      const float val = oacc[c][j] * rl[j];
      Ag[((size_t)b * Sc + s) * Hc + head * HDc + c * 16 + col] = (__bf16)val;
    }
  }
}

// ---------------------------------------------------------------------------
extern "C" void kernel_launch(void* const* d_in, const int* in_sizes, int n_in,
                              void* d_out, int out_size, void* d_ws,
                              size_t ws_size, hipStream_t stream) {
  const float* hs = (const float*)d_in[0];
  const float* wq = (const float*)d_in[1];
  const float* bq = (const float*)d_in[2];
  const float* wk = (const float*)d_in[3];
  const float* bk = (const float*)d_in[4];
  const float* wv = (const float*)d_in[5];
  const float* bv = (const float*)d_in[6];
  const float* wo = (const float*)d_in[7];
  const float* bo = (const float*)d_in[8];
  float* out = (float*)d_out;

  const size_t per = (size_t)Bc * NHc * Sc * HDc;  // 8,388,608 elements
  float* ws = (float*)d_ws;
  float* qf = ws;                 // f32 Q [B,NH,S,HD]
  float* kf = ws + per;           // f32 K
  float* vf = ws + 2 * per;       // f32 V
  __bf16* hb  = (__bf16*)(ws + 3 * per);  // bf16 hidden [B*S, H]
  __bf16* wqb = hb + per;                 // bf16 weights [H, H]
  __bf16* wkb = wqb + per / 2;
  __bf16* wvb = wkb + per / 2;
  __bf16* wob = wvb + per / 2;
  __bf16* qb  = wob + per / 2;            // bf16 Q (RoPE+scale)
  __bf16* kb  = qb + per;                 // bf16 K (RoPE)
  __bf16* vtb = kb + per;                 // bf16 V^T [B,NH,HD,S]
  __bf16* ab  = vtb + per;                // bf16 attn out [B,S,H]

  const dim3 blk(128);
  const dim3 pgrid(Hc / 128, (Bc * Sc) / 64);  // (16, 64)

  // 1) one-time bf16 packs
  pack_bf16_kernel<<<(int)(per / 1024), 256, 0, stream>>>(hs, hb, per);
  pack_bf16_kernel<<<(int)(per / 2048), 256, 0, stream>>>(wq, wqb, per / 2);
  pack_bf16_kernel<<<(int)(per / 2048), 256, 0, stream>>>(wk, wkb, per / 2);
  pack_bf16_kernel<<<(int)(per / 2048), 256, 0, stream>>>(wv, wvb, per / 2);
  pack_bf16_kernel<<<(int)(per / 2048), 256, 0, stream>>>(wo, wob, per / 2);

  // 2) QKV projections (f32 out, [B,NH,S,HD])
  gemm_bf16_kernel<<<pgrid, blk, 0, stream>>>(hb, wqb, bq, qf, Bc * Sc, Hc, Hc, 1);
  gemm_bf16_kernel<<<pgrid, blk, 0, stream>>>(hb, wkb, bk, kf, Bc * Sc, Hc, Hc, 1);
  gemm_bf16_kernel<<<pgrid, blk, 0, stream>>>(hb, wvb, bv, vf, Bc * Sc, Hc, Hc, 1);

  // 3) RoPE + pack (scale folded into Q); V transpose-pack
  const int rope_pairs = Bc * NHc * Sc * 64;
  const float scale = 0.08838834764831845f;  // 1/sqrt(128)
  rope_pack_kernel<<<rope_pairs / 256, 256, 0, stream>>>(qf, qb, rope_pairs, scale);
  rope_pack_kernel<<<rope_pairs / 256, 256, 0, stream>>>(kf, kb, rope_pairs, 1.0f);
  packT_kernel<<<(int)(per / 256), 256, 0, stream>>>(vf, vtb, per);

  // 4) flash attention -> bf16 [B,S,H]
  attn_kernel<<<dim3(Sc / 64, Bc * NHc), blk, 0, stream>>>(qb, kb, vtb, ab);

  // 5) output projection -> f32 d_out
  gemm_bf16_kernel<<<pgrid, blk, 0, stream>>>(ab, wob, bo, out, Bc * Sc, Hc, Hc, 0);

  (void)in_sizes; (void)n_in; (void)out_size; (void)ws_size;
}
